// BitBottleneck_75385265979773
// MI455X (gfx1250) — matrix operand order repaired and checked by reading, not compiled, not run on hardware
//
#include <hip/hip_runtime.h>
#include <hip/hip_bf16.h>

typedef __bf16 bf16;
typedef __attribute__((ext_vector_type(8)))  __bf16 v8bf;
typedef __attribute__((ext_vector_type(16))) __bf16 v16bf;
typedef __attribute__((ext_vector_type(8)))  float  v8f;

#define NBINS 8192

// ---------------- quantization pipeline ----------------

__global__ void zero_u32(uint32_t* p, int n) {
  int i = blockIdx.x * blockDim.x + threadIdx.x;
  int st = gridDim.x * blockDim.x;
  for (; i < n; i += st) p[i] = 0u;
}

__global__ void absmax_k(const float* __restrict__ w, int n, uint32_t* mx) {
  int i = blockIdx.x * blockDim.x + threadIdx.x;
  int st = gridDim.x * blockDim.x;
  uint32_t loc = 0u;
  for (; i < n; i += st) {
    uint32_t b = __float_as_uint(fabsf(w[i]));
    loc = loc > b ? loc : b;
  }
  atomicMax(mx, loc);
}

__global__ void hist_k(const float* __restrict__ w, int n,
                       const uint32_t* __restrict__ mxp, uint32_t* __restrict__ hist) {
  float mx = __uint_as_float(*mxp);
  float scale = (mx > 0.f) ? ((float)NBINS / mx) : 0.f;
  int i = blockIdx.x * blockDim.x + threadIdx.x;
  int st = gridDim.x * blockDim.x;
  for (; i < n; i += st) {
    int b = (int)(fabsf(w[i]) * scale);
    b = b < (NBINS - 1) ? b : (NBINS - 1);
    atomicAdd(&hist[b], 1u);
  }
}

__global__ void median_k(const uint32_t* __restrict__ hist,
                         const uint32_t* __restrict__ mxp,
                         float* __restrict__ s_out, int n) {
  if (threadIdx.x != 0 || blockIdx.x != 0) return;
  uint32_t half = ((uint32_t)n + 1u) / 2u;
  uint32_t cum = 0u;
  int bin = NBINS - 1;
  for (int i = 0; i < NBINS; ++i) {
    cum += hist[i];
    if (cum >= half) { bin = i; break; }
  }
  float mx = __uint_as_float(*mxp);
  float s = ((float)bin + 0.5f) * mx / (float)NBINS;
  *s_out = fmaxf(s, 1e-5f);
}

// ternary-quantize and permute OIHW -> [Co][tap][Ci] bf16
__global__ void quant_perm_k(const float* __restrict__ w, const float* __restrict__ sp,
                             bf16* __restrict__ wq, int n, int CIp) {
  float s = *sp;
  float rs = 1.f / s;
  int i = blockIdx.x * blockDim.x + threadIdx.x;
  int st = gridDim.x * blockDim.x;
  for (; i < n; i += st) {
    int r  = i % 9;             // tap
    int t9 = i / 9;             // co*CI + ci
    int ci = t9 % CIp;
    int co = t9 / CIp;
    float q = rintf(w[i] * rs);
    q = fminf(fmaxf(q, -1.f), 1.f) * s;
    wq[((size_t)co * 9 + r) * CIp + ci] = (bf16)q;
  }
}

// x: fp32 NCHW [32][256][3136] -> bf16 NHWC-flat [p][256]
__global__ void cvt_nhwc_k(const float* __restrict__ x, bf16* __restrict__ xb, int n) {
  int i = blockIdx.x * blockDim.x + threadIdx.x;
  int st = gridDim.x * blockDim.x;
  for (; i < n; i += st) {
    int ci = i & 255;           // CI = 256
    int p  = i >> 8;
    int im = p / 3136;
    int hw = p - im * 3136;
    xb[i] = (bf16)x[((size_t)im * 256 + ci) * 3136 + hw];
  }
}

// ---------------- implicit-GEMM conv3x3 + BN + SiLU (+residual) ----------------
//
// K ordered as (tap, ci). Block: 128(M) x 128(N); 8 waves as 4(M) x 2(N),
// each wave computes 32x64 via 8 accumulators (8 v_wmma per 32-K step).
// Global->LDS staging uses global_load_async_to_lds_b128 (ASYNCcnt) with
// two LDS stages -> single barrier per K-step.

__device__ __forceinline__ v16bf frag2(const bf16* row, int o0, int o1) {
  v8bf lo = *(const v8bf*)(row + o0);
  v8bf hi = *(const v8bf*)(row + o1);
  return __builtin_shufflevector(lo, hi, 0,1,2,3,4,5,6,7,8,9,10,11,12,13,14,15);
}

template<int CI, int CO, bool RESID>
__global__ __launch_bounds__(256) void conv_wmma(
    const bf16* __restrict__ xin,  // [P][CI] bf16 (NHWC-flat)
    const bf16* __restrict__ wq,   // [CO][9][CI] bf16 quantized
    const float* __restrict__ gam, const float* __restrict__ bet,
    const float* __restrict__ mean, const float* __restrict__ var,
    const float* __restrict__ resid,   // fp32 NCHW residual (RESID only)
    float* __restrict__ outf,          // fp32 NCHW out (RESID only)
    bf16* __restrict__ outb)           // bf16 NHWC out (!RESID)
{
  constexpr int K   = 9 * CI;
  constexpr int HW  = 3136;
  constexpr int SPT = CI / 32;          // K-stages per tap (8 or 4, pow2)
  constexpr int S   = 9 * SPT;          // total K-stages
  constexpr uint32_t STAGE_B = 256 * 32 * 2;  // bytes per LDS stage

  // stage layout: rows 0..127 = A (pixels), rows 128..255 = B (out-channels)
  __shared__ __align__(16) bf16 smem[2][256][32];

  const int t  = threadIdx.x;
  const int m0 = blockIdx.x * 128;  // pixel tile base
  const int n0 = blockIdx.y * 128;  // channel tile base

  // ---- A loader: thread -> (pixel row, 16-ci chunk) ----
  const int tm     = t >> 1;            // 0..127
  const int cchunk = (t & 1) * 16;
  const int ap  = m0 + tm;              // flat pixel
  const int ahw = ap % HW;
  const int ah  = ahw / 56;
  const int aw  = ahw % 56;

  // ---- B loader: thread -> (channel row, 16-k chunk) ----
  const int bn2    = t >> 1;            // 0..127
  const int bchunk = (t & 1) * 16;

  // ---- wave geometry: wave computes 32(M) x 64(N) ----
  const int lane = t & 31, wid = t >> 5;
  const int wm = wid & 3, wn = wid >> 2;
  const int lh = lane >> 4, ll = lane & 15;
  const int am0   = wm * 32 + ll;       // A fragment rows am0, am0+16
  const int abase = lh * 8;             // A K-offset per ISA layout
  const int koff  = lh * 8;             // B K-offset per ISA layout

  v8f acc[2][4];
#pragma unroll
  for (int i = 0; i < 2; ++i)
#pragma unroll
    for (int j = 0; j < 4; ++j)
#pragma unroll
      for (int e = 0; e < 8; ++e) acc[i][j][e] = 0.f;

  v8bf zero8;
#pragma unroll
  for (int e = 0; e < 8; ++e) zero8[e] = (bf16)0.0f;

  const bf16* lds0 = &smem[0][0][0];

  // issue async global->LDS copies for K-stage s into buffer s&1
  auto issue = [&](int s) {
    const int tap = s / SPT;
    const int c0  = (s - tap * SPT) * 32;
    const int buf = s & 1;
    const int t3  = tap / 3;
    const int kh  = t3 - 1;
    const int kw  = (tap - t3 * 3) - 1;
    const bool valid = ((unsigned)(ah + kh) < 56u) & ((unsigned)(aw + kw) < 56u);

    const uint32_t ldsA = (uint32_t)((tm * 32 + cchunk) * 2) + (uint32_t)buf * STAGE_B;
    if (valid) {
      const uint32_t ga = (uint32_t)(((ap + kh * 56 + kw) * CI + c0 + cchunk) * 2);
      asm volatile("global_load_async_to_lds_b128 %0, %1, %2"
                   :: "v"(ldsA), "v"(ga), "s"(xin) : "memory");
      asm volatile("global_load_async_to_lds_b128 %0, %1, %2 offset:16"
                   :: "v"(ldsA), "v"(ga), "s"(xin) : "memory");
    } else if (c0 < 64) {
      // padding rows: zero each buffer once per tap (sole writer of these cells)
      *(v8bf*)&smem[buf][tm][cchunk]     = zero8;
      *(v8bf*)&smem[buf][tm][cchunk + 8] = zero8;
    }

    const uint32_t ldsB = (uint32_t)(((128 + bn2) * 32 + bchunk) * 2) + (uint32_t)buf * STAGE_B;
    const uint32_t gb = (uint32_t)((((n0 + bn2) * K) + tap * CI + c0 + bchunk) * 2);
    asm volatile("global_load_async_to_lds_b128 %0, %1, %2"
                 :: "v"(ldsB), "v"(gb), "s"(wq) : "memory");
    asm volatile("global_load_async_to_lds_b128 %0, %1, %2 offset:16"
                 :: "v"(ldsB), "v"(gb), "s"(wq) : "memory");
  };

  // WMMA on K-stage s from buffer s&1
  auto compute = [&](int s) {
    const bf16* base = lds0 + (size_t)(s & 1) * (256 * 32);
    v16bf a0 = frag2(base + (size_t)am0 * 32,        abase, abase + 16);
    v16bf a1 = frag2(base + (size_t)(am0 + 16) * 32, abase, abase + 16);
#pragma unroll
    for (int q = 0; q < 4; ++q) {
      const int brow = 128 + wn * 64 + q * 16 + ll;
      v16bf bq = frag2(base + (size_t)brow * 32, koff, koff + 16);
      acc[0][q] = __builtin_amdgcn_wmma_f32_16x16x32_bf16(false, a0, false, bq, (short)0, acc[0][q], false, false);
      acc[1][q] = __builtin_amdgcn_wmma_f32_16x16x32_bf16(false, a1, false, bq, (short)0, acc[1][q], false, false);
    }
  };

  issue(0);
  asm volatile("s_wait_asynccnt 0x0" ::: "memory");
  __syncthreads();

#pragma unroll 1
  for (int s = 0; s < S; ++s) {
    if (s + 1 < S) issue(s + 1);
    compute(s);
    asm volatile("s_wait_asynccnt 0x0" ::: "memory");
    __syncthreads();
  }

  // ---- fused BN + SiLU epilogue (+ residual) ----
#pragma unroll
  for (int q = 0; q < 4; ++q) {
    const int co = n0 + wn * 64 + q * 16 + ll;
    const float inv  = gam[co] * rsqrtf(var[co] + 1e-5f);
    const float bias = bet[co] - mean[co] * inv;
#pragma unroll
    for (int mi = 0; mi < 2; ++mi) {
      const int p0 = m0 + wm * 32 + mi * 16 + lh * 8;  // 8 consecutive pixels
      if (RESID) {
        const int im  = p0 / HW;
        const int hwb = p0 - im * HW;
        const size_t base = ((size_t)im * CO + co) * HW + hwb;
#pragma unroll
        for (int r = 0; r < 8; ++r) {
          float y = acc[mi][q][r] * inv + bias;
          y = y / (1.f + __expf(-y));
          float rr = __builtin_nontemporal_load(&resid[base + r]);
          __builtin_nontemporal_store(rr + y, &outf[base + r]);
        }
      } else {
        const size_t base = (size_t)p0 * CO + co;
#pragma unroll
        for (int r = 0; r < 8; ++r) {
          float y = acc[mi][q][r] * inv + bias;
          y = y / (1.f + __expf(-y));
          outb[base + (size_t)r * CO] = (bf16)y;
        }
      }
    }
  }
}

// ---------------- launcher ----------------

extern "C" void kernel_launch(void* const* d_in, const int* in_sizes, int n_in,
                              void* d_out, int out_size, void* d_ws, size_t ws_size,
                              hipStream_t stream) {
  const float* x  = (const float*)d_in[0];
  const float* w1 = (const float*)d_in[1];
  const float* g1 = (const float*)d_in[2];
  const float* b1 = (const float*)d_in[3];
  const float* m1 = (const float*)d_in[4];
  const float* v1 = (const float*)d_in[5];
  const float* w2 = (const float*)d_in[6];
  const float* g2 = (const float*)d_in[7];
  const float* b2 = (const float*)d_in[8];
  const float* m2 = (const float*)d_in[9];
  const float* v2 = (const float*)d_in[10];

  const int NW = 128 * 256 * 9;          // 294912
  const int NX = 32 * 256 * 3136;        // 25690112

  char* ws = (char*)d_ws;
  uint32_t* hist1 = (uint32_t*)(ws + 0);
  uint32_t* hist2 = (uint32_t*)(ws + 32768);
  uint32_t* scal  = (uint32_t*)(ws + 65536);   // [0]=max1 [1]=max2 [2]=s1 [3]=s2
  float*    s1p   = (float*)(scal + 2);
  float*    s2p   = (float*)(scal + 3);
  bf16* wq1 = (bf16*)(ws + 131072);                         // [128][9][256]
  bf16* wq2 = (bf16*)(ws + 720896);                         // [256][9][128]
  bf16* xb  = (bf16*)(ws + 1310720);                        // [P][256] NHWC-flat
  bf16* hb  = (bf16*)(ws + 1310720 + (size_t)NX * 2);       // [P][128] NHWC-flat

  zero_u32<<<64, 256, 0, stream>>>((uint32_t*)ws, 16448);

  absmax_k<<<576, 256, 0, stream>>>(w1, NW, &scal[0]);
  absmax_k<<<576, 256, 0, stream>>>(w2, NW, &scal[1]);

  hist_k<<<576, 256, 0, stream>>>(w1, NW, &scal[0], hist1);
  hist_k<<<576, 256, 0, stream>>>(w2, NW, &scal[1], hist2);

  median_k<<<1, 32, 0, stream>>>(hist1, &scal[0], s1p, NW);
  median_k<<<1, 32, 0, stream>>>(hist2, &scal[1], s2p, NW);

  quant_perm_k<<<576, 256, 0, stream>>>(w1, s1p, wq1, NW, 256);
  quant_perm_k<<<576, 256, 0, stream>>>(w2, s2p, wq2, NW, 128);

  cvt_nhwc_k<<<4096, 256, 0, stream>>>(x, xb, NX);

  // conv1: 256 -> 128, out h (bf16 NHWC-flat)
  conv_wmma<256, 128, false><<<dim3(784, 1), 256, 0, stream>>>(
      xb, wq1, g1, b1, m1, v1, nullptr, nullptr, hb);

  // conv2: 128 -> 256, + BN + SiLU + residual, out fp32 NCHW
  conv_wmma<128, 256, true><<<dim3(784, 2), 256, 0, stream>>>(
      hb, wq2, g2, b2, m2, v2, x, (float*)d_out, nullptr);
}
